// TransformerEncoderBlock_24661702214161
// MI455X (gfx1250) — compile-verified, matrix operand order checked by reading, not run on hardware
//
#include <hip/hip_runtime.h>
#include <hip/hip_bf16.h>
#include <math.h>

// Problem constants (match reference)
#define BB  8
#define SS  1024
#define DD  1024
#define HH  16
#define DHH 64

typedef __attribute__((ext_vector_type(16))) __bf16 v16bf;
typedef __attribute__((ext_vector_type(8)))  float  v8f;

union FragBF {
  uint4 u[2];
  v16bf v;
};

__device__ inline v8f v8f_zero() {
  v8f z = {0.f, 0.f, 0.f, 0.f, 0.f, 0.f, 0.f, 0.f};
  return z;
}

// Async 16-B copy global -> LDS (CDNA5 GLOBAL_LOAD_ASYNC_TO_LDS_B128).
// vdst = per-lane LDS byte address, vaddr = per-lane 64-bit global address.
// Tracked by ASYNCcnt; drained with s_wait_asynccnt.
__device__ inline void async_copy_b128(const __bf16* gptr, __bf16* lptr) {
  const unsigned loff = (unsigned)(uintptr_t)lptr;  // low 32b of flat = LDS addr
  asm volatile("global_load_async_to_lds_b128 %0, %1, off"
               :
               : "v"(loff), "v"(gptr)
               : "memory");
}

// ---------------------------------------------------------------------------
// Generic bf16 GEMM:  out[M,N] = A[M,K] * W[N,K]^T + bias
// 128x128 block tile, BK=32, 256 threads (8 waves), each wave 2x4 WMMA tiles.
// Double-buffered LDS fed by async global->LDS copies.
// ---------------------------------------------------------------------------
template <bool OUT_BF16>
__global__ __launch_bounds__(256, 2)
void gemm_bf16_kernel(const __bf16* __restrict__ A,    // [M,K] row-major
                      const __bf16* __restrict__ W,    // [N,K] row-major
                      const float*  __restrict__ bias, // [N] or nullptr
                      void* __restrict__ outp,
                      int M, int N, int K)
{
  constexpr int LDT = 40;  // 32 + 8 bf16 pad; 80 B row stride (16-B aligned)
  __shared__ __align__(16) __bf16 As[2][128 * LDT];
  __shared__ __align__(16) __bf16 Ws[2][128 * LDT];

  const int  tid  = threadIdx.x;
  const int  lane = tid & 31;
  const int  wave = tid >> 5;
  const int  wm   = wave >> 1;  // 0..3 : 32-row slab
  const int  wn   = wave & 1;   // 0..1 : 64-col slab
  const long bm   = (long)blockIdx.y * 128;
  const long bn   = (long)blockIdx.x * 128;

  const int fr = lane & 15;         // fragment row/col within 16
  const int kb = (lane >> 4) * 8;   // fragment K base (A-layout halves)

  v8f acc[2][4];
  for (int i = 0; i < 2; ++i)
    for (int j = 0; j < 4; ++j) acc[i][j] = v8f_zero();

  const int lrow = tid >> 1;        // 0..127
  const int lcol = (tid & 1) * 16;  // 0 or 16 (bf16 elems)

  const __bf16* gA = A + (bm + lrow) * (long)K + lcol;
  const __bf16* gW = W + (bn + lrow) * (long)K + lcol;

  auto issue_stage = [&](int k0, int buf) {
    async_copy_b128(gA + k0,     &As[buf][lrow * LDT + lcol]);
    async_copy_b128(gA + k0 + 8, &As[buf][lrow * LDT + lcol + 8]);
    async_copy_b128(gW + k0,     &Ws[buf][lrow * LDT + lcol]);
    async_copy_b128(gW + k0 + 8, &Ws[buf][lrow * LDT + lcol + 8]);
  };

  issue_stage(0, 0);   // prologue
  int cur = 0;

  for (int k0 = 0; k0 < K; k0 += 32) {
    const bool has_next = (k0 + 32) < K;
    if (has_next) {
      // Buffer cur^1 was last read in iteration k0-32 and a barrier has
      // occurred since, so it is safe to overwrite.
      issue_stage(k0 + 32, cur ^ 1);
      asm volatile("s_wait_asynccnt 0x4" ::: "memory");  // current 4 landed
    } else {
      asm volatile("s_wait_asynccnt 0x0" ::: "memory");
    }
    __syncthreads();  // everyone's current-stage tiles visible

    const __bf16* as = As[cur];
    const __bf16* wsld = Ws[cur];

    FragBF af[2];
    for (int i = 0; i < 2; ++i) {
      const int r = wm * 32 + i * 16 + fr;
      af[i].u[0] = *reinterpret_cast<const uint4*>(&as[r * LDT + kb]);
      af[i].u[1] = *reinterpret_cast<const uint4*>(&as[r * LDT + kb + 16]);
    }
    for (int j = 0; j < 4; ++j) {
      FragBF wf;
      const int c = wn * 64 + j * 16 + fr;
      wf.u[0] = *reinterpret_cast<const uint4*>(&wsld[c * LDT + kb]);
      wf.u[1] = *reinterpret_cast<const uint4*>(&wsld[c * LDT + kb + 16]);
      for (int i = 0; i < 2; ++i) {
        acc[i][j] = __builtin_amdgcn_wmma_f32_16x16x32_bf16(
            false, af[i].v, false, wf.v, (short)0, acc[i][j], false, false);
      }
    }
    __syncthreads();  // all reads of buf cur done before it is refilled
    cur ^= 1;
  }

  for (int i = 0; i < 2; ++i) {
    const long row0 = bm + wm * 32 + i * 16 + (lane >> 4) * 8;
    for (int j = 0; j < 4; ++j) {
      const long col = bn + wn * 64 + j * 16 + (lane & 15);
      const float bv = bias ? bias[col] : 0.0f;
      for (int r = 0; r < 8; ++r) {
        const float v = acc[i][j][r] + bv;
        if constexpr (OUT_BF16) {
          reinterpret_cast<__bf16*>(outp)[(row0 + r) * (long)N + col] = (__bf16)v;
        } else {
          reinterpret_cast<float*>(outp)[(row0 + r) * (long)N + col] = v;
        }
      }
    }
  }
}

// ---------------------------------------------------------------------------
// Flash attention: causal, scale 1/sqrt(64), query-dim padding mask.
// Block = 128 threads (4 waves); each wave owns 16 query rows; block covers
// 64 query rows of one (b,h). K/V tiles of 64 keys staged in LDS.
// Output written merged-head bf16 [B*S, D].
// ---------------------------------------------------------------------------
__global__ __launch_bounds__(128, 2)
void flash_attn_kernel(const __bf16* __restrict__ Q,
                       const __bf16* __restrict__ Kk,
                       const __bf16* __restrict__ V,
                       const int* __restrict__ pad,
                       __bf16* __restrict__ O)
{
  constexpr int LDT = 72;                           // 64 + 8 pad, 144 B stride
  __shared__ __align__(16) __bf16 Kt[64 * LDT];     // [key][d]
  __shared__ __align__(16) __bf16 Vt[64 * LDT];     // [d][key] (transposed)
  __shared__ __align__(16) __bf16 Pst[4][16 * LDT]; // per-wave P staging

  const int tid  = threadIdx.x;
  const int lane = tid & 31;
  const int wave = tid >> 5;              // 0..3
  const int bh   = blockIdx.y;
  const int b    = bh >> 4;               // / HH
  const int h    = bh & 15;
  const int qb   = blockIdx.x * 64;

  const int fr = lane & 15;
  const int kb = (lane >> 4) * 8;

  // Q fragments for this wave's 16 rows (K-dim = 64 -> two 16x32 frags)
  const long qrow = (long)b * SS + qb + wave * 16 + fr;
  const __bf16* qp = Q + qrow * DD + h * DHH;
  FragBF qf[2];
  qf[0].u[0] = *reinterpret_cast<const uint4*>(qp + kb);
  qf[0].u[1] = *reinterpret_cast<const uint4*>(qp + kb + 16);
  qf[1].u[0] = *reinterpret_cast<const uint4*>(qp + kb + 32);
  qf[1].u[1] = *reinterpret_cast<const uint4*>(qp + kb + 48);

  const int mrow0 = qb + wave * 16 + (lane >> 4) * 8;  // C-layout row base (+r)
  int padok[8];
  for (int r = 0; r < 8; ++r) padok[r] = pad[(long)b * SS + mrow0 + r];

  float mrun[8], lrun[8];
  for (int r = 0; r < 8; ++r) { mrun[r] = -INFINITY; lrun[r] = 0.0f; }
  v8f o[4];
  for (int t = 0; t < 4; ++t) o[t] = v8f_zero();

  const int ntiles = blockIdx.x + 1;      // causal bound
  for (int kt = 0; kt < ntiles; ++kt) {
    const int kbase = kt * 64;
    __syncthreads();
    {
      const int krow = tid >> 1;          // 0..63
      const int dc   = (tid & 1) * 32;    // 0 or 32
      const uint4* kp = reinterpret_cast<const uint4*>(
          Kk + ((long)b * SS + kbase + krow) * DD + h * DHH + dc);
      uint4 kr[4] = {kp[0], kp[1], kp[2], kp[3]};
      uint4* kd = reinterpret_cast<uint4*>(&Kt[krow * LDT + dc]);
      kd[0] = kr[0]; kd[1] = kr[1]; kd[2] = kr[2]; kd[3] = kr[3];

      const uint4* vp = reinterpret_cast<const uint4*>(
          V + ((long)b * SS + kbase + krow) * DD + h * DHH + dc);
      uint4 vr[4] = {vp[0], vp[1], vp[2], vp[3]};
      const __bf16* tv = reinterpret_cast<const __bf16*>(vr);
      for (int e = 0; e < 32; ++e)
        Vt[(dc + e) * LDT + krow] = tv[e];
    }
    __syncthreads();

    // S = Q * K^T : 16 x 64 (4 output tiles, 2 k-steps each)
    v8f s[4];
    for (int t = 0; t < 4; ++t) s[t] = v8f_zero();
    for (int t = 0; t < 4; ++t) {
      for (int ks = 0; ks < 2; ++ks) {
        FragBF kf;
        const int krow = t * 16 + fr;
        kf.u[0] = *reinterpret_cast<const uint4*>(&Kt[krow * LDT + ks * 32 + kb]);
        kf.u[1] = *reinterpret_cast<const uint4*>(&Kt[krow * LDT + ks * 32 + kb + 16]);
        s[t] = __builtin_amdgcn_wmma_f32_16x16x32_bf16(
            false, qf[ks].v, false, kf.v, (short)0, s[t], false, false);
      }
    }

    // scale + masks (element (m,n): row = mrow0 + r, col = kbase + t*16 + n)
    for (int t = 0; t < 4; ++t) {
      const int kcol = kbase + t * 16 + (lane & 15);
      for (int r = 0; r < 8; ++r) {
        float e = s[t][r] * 0.125f;         // 1/sqrt(64)
        if (!padok[r]) e = -1.0e9f;         // query-dim padding mask
        if (kcol > mrow0 + r) e = -3.0e38f; // causal
        s[t][r] = e;
      }
    }

    // online softmax (row reductions across the 16-lane halves)
    for (int r = 0; r < 8; ++r) {
      float rm = -3.0e38f;
      for (int t = 0; t < 4; ++t) rm = fmaxf(rm, s[t][r]);
      for (int off = 1; off < 16; off <<= 1)
        rm = fmaxf(rm, __shfl_xor(rm, off, 32));
      const float mn    = fmaxf(mrun[r], rm);
      const float alpha = __expf(mrun[r] - mn);
      mrun[r] = mn;
      lrun[r] *= alpha;
      for (int t = 0; t < 4; ++t) o[t][r] *= alpha;
      float rs = 0.0f;
      for (int t = 0; t < 4; ++t) {
        const float p = __expf(s[t][r] - mn);
        s[t][r] = p;
        rs += p;
      }
      for (int off = 1; off < 16; off <<= 1)
        rs += __shfl_xor(rs, off, 32);
      lrun[r] += rs;
    }

    // stage P (C-layout -> A-layout transpose through per-wave LDS)
    for (int t = 0; t < 4; ++t)
      for (int r = 0; r < 8; ++r)
        Pst[wave][((lane >> 4) * 8 + r) * LDT + t * 16 + (lane & 15)] =
            (__bf16)s[t][r];
    asm volatile("s_wait_dscnt 0" ::: "memory");

    FragBF pf[2];
    for (int ks = 0; ks < 2; ++ks) {
      pf[ks].u[0] = *reinterpret_cast<const uint4*>(&Pst[wave][fr * LDT + ks * 32 + kb]);
      pf[ks].u[1] = *reinterpret_cast<const uint4*>(&Pst[wave][fr * LDT + ks * 32 + kb + 16]);
    }
    for (int td = 0; td < 4; ++td) {
      for (int ks = 0; ks < 2; ++ks) {
        FragBF vf;
        const int vrow = td * 16 + fr;
        vf.u[0] = *reinterpret_cast<const uint4*>(&Vt[vrow * LDT + ks * 32 + kb]);
        vf.u[1] = *reinterpret_cast<const uint4*>(&Vt[vrow * LDT + ks * 32 + kb + 16]);
        o[td] = __builtin_amdgcn_wmma_f32_16x16x32_bf16(
            false, pf[ks].v, false, vf.v, (short)0, o[td], false, false);
      }
    }
  }

  // normalize and store merged-head bf16 output
  for (int td = 0; td < 4; ++td) {
    const long orow = (long)b * SS + mrow0;
    const int  ocol = h * DHH + td * 16 + (lane & 15);
    for (int r = 0; r < 8; ++r) {
      const float v = o[td][r] / lrun[r];
      O[(orow + r) * DD + ocol] = (__bf16)v;
    }
  }
}

// ---------------------------------------------------------------------------
// Fused residual add + LayerNorm over D=1024 (one row per block).
// Writes f32 always; optional bf16 side output for the next GEMM.
// ---------------------------------------------------------------------------
__global__ __launch_bounds__(256)
void add_layernorm_kernel(const float* __restrict__ X,
                          const float* __restrict__ Y,
                          const float* __restrict__ w,
                          const float* __restrict__ bcoef,
                          float* __restrict__ outF,
                          __bf16* __restrict__ outB)
{
  __shared__ float rs[256], rs2[256];
  const long row = blockIdx.x;
  const int  t   = threadIdx.x;
  float loc[4];
  float s = 0.f, s2 = 0.f;
  for (int i = 0; i < 4; ++i) {
    const int c = t * 4 + i;
    const float v = X[row * DD + c] + Y[row * DD + c];
    loc[i] = v; s += v; s2 += v * v;
  }
  rs[t] = s; rs2[t] = s2;
  __syncthreads();
  for (int off = 128; off > 0; off >>= 1) {
    if (t < off) { rs[t] += rs[t + off]; rs2[t] += rs2[t + off]; }
    __syncthreads();
  }
  const float mean = rs[0] * (1.0f / DD);
  const float var  = rs2[0] * (1.0f / DD) - mean * mean;
  const float rstd = rsqrtf(var + 1e-12f);
  for (int i = 0; i < 4; ++i) {
    const int c = t * 4 + i;
    const float v = w[c] * ((loc[i] - mean) * rstd) + bcoef[c];
    outF[row * DD + c] = v;
    if (outB) outB[row * DD + c] = (__bf16)v;
  }
}

// ---------------------------------------------------------------------------
// tanh-GELU (f32 in -> bf16 out) and f32->bf16 conversion
// ---------------------------------------------------------------------------
__global__ __launch_bounds__(256)
void gelu_bf16_kernel(const float* __restrict__ in, __bf16* __restrict__ out, long n)
{
  const long i0 = ((long)blockIdx.x * 256 + threadIdx.x) * 4;
  for (int i = 0; i < 4; ++i) {
    const long idx = i0 + i;
    if (idx < n) {
      const float h = in[idx];
      const float u = 0.7978845608028654f * (h + 0.044715f * h * h * h);
      out[idx] = (__bf16)(0.5f * h * (1.0f + tanhf(u)));
    }
  }
}

__global__ __launch_bounds__(256)
void f32_to_bf16_kernel(const float* __restrict__ in, __bf16* __restrict__ out, long n)
{
  const long i0 = ((long)blockIdx.x * 256 + threadIdx.x) * 4;
  for (int i = 0; i < 4; ++i) {
    const long idx = i0 + i;
    if (idx < n) out[idx] = (__bf16)in[idx];
  }
}

// ---------------------------------------------------------------------------
extern "C" void kernel_launch(void* const* d_in, const int* in_sizes, int n_in,
                              void* d_out, int out_size, void* d_ws, size_t ws_size,
                              hipStream_t stream)
{
  (void)in_sizes; (void)n_in; (void)out_size; (void)ws_size;

  const float* hidden = (const float*)d_in[0];
  const int*   pad    = (const int*)d_in[1];
  const float* wq = (const float*)d_in[2];   const float* bq = (const float*)d_in[3];
  const float* wk = (const float*)d_in[4];   const float* bk = (const float*)d_in[5];
  const float* wv = (const float*)d_in[6];   const float* bv = (const float*)d_in[7];
  const float* wo = (const float*)d_in[8];   const float* bo = (const float*)d_in[9];
  const float* w1 = (const float*)d_in[10];  const float* b1 = (const float*)d_in[11];
  const float* w2 = (const float*)d_in[12];  const float* b2 = (const float*)d_in[13];
  const float* ln1w = (const float*)d_in[14]; const float* ln1b = (const float*)d_in[15];
  const float* ln2w = (const float*)d_in[16]; const float* ln2b = (const float*)d_in[17];
  float* outp = (float*)d_out;

  char* ws = (char*)d_ws;
  size_t off = 0;
  auto carve = [&](size_t bytes) -> char* {
    char* p = ws + off;
    off = (off + bytes + 255) & ~(size_t)255;
    return p;
  };
  const long MT = (long)BB * SS;  // 8192 rows

  __bf16* x_bf   = (__bf16*)carve((size_t)MT * DD * 2);
  __bf16* wq_bf  = (__bf16*)carve((size_t)DD * DD * 2);
  __bf16* wk_bf  = (__bf16*)carve((size_t)DD * DD * 2);
  __bf16* wv_bf  = (__bf16*)carve((size_t)DD * DD * 2);
  __bf16* wo_bf  = (__bf16*)carve((size_t)DD * DD * 2);
  __bf16* w1_bf  = (__bf16*)carve((size_t)4 * DD * DD * 2);
  __bf16* w2_bf  = (__bf16*)carve((size_t)4 * DD * DD * 2);
  __bf16* q_bf   = (__bf16*)carve((size_t)MT * DD * 2);
  __bf16* k_bf   = (__bf16*)carve((size_t)MT * DD * 2);
  __bf16* v_bf   = (__bf16*)carve((size_t)MT * DD * 2);
  __bf16* at_bf  = (__bf16*)carve((size_t)MT * DD * 2);
  float*  proj_f = (float*) carve((size_t)MT * DD * 4);
  float*  h_f    = (float*) carve((size_t)MT * DD * 4);
  __bf16* h_bf   = (__bf16*)carve((size_t)MT * DD * 2);
  float*  ff1_f  = (float*) carve((size_t)MT * 4 * DD * 4);
  __bf16* ff1_bf = (__bf16*)carve((size_t)MT * 4 * DD * 2);
  float*  ff2_f  = (float*) carve((size_t)MT * DD * 4);

  auto cvt = [&](const float* src, __bf16* dst, long n) {
    f32_to_bf16_kernel<<<dim3((unsigned)((n + 1023) / 1024)), 256, 0, stream>>>(src, dst, n);
  };

  // 1) conversions to bf16
  cvt(hidden, x_bf, MT * DD);
  cvt(wq, wq_bf, (long)DD * DD);
  cvt(wk, wk_bf, (long)DD * DD);
  cvt(wv, wv_bf, (long)DD * DD);
  cvt(wo, wo_bf, (long)DD * DD);
  cvt(w1, w1_bf, (long)4 * DD * DD);
  cvt(w2, w2_bf, (long)4 * DD * DD);

  // 2) QKV projections (bf16 out, merged-head row-major [B*S, D])
  const dim3 gDD(DD / 128, (unsigned)(MT / 128));
  gemm_bf16_kernel<true><<<gDD, 256, 0, stream>>>(x_bf, wq_bf, bq, q_bf, (int)MT, DD, DD);
  gemm_bf16_kernel<true><<<gDD, 256, 0, stream>>>(x_bf, wk_bf, bk, k_bf, (int)MT, DD, DD);
  gemm_bf16_kernel<true><<<gDD, 256, 0, stream>>>(x_bf, wv_bf, bv, v_bf, (int)MT, DD, DD);

  // 3) causal flash attention -> at_bf [B*S, D]
  flash_attn_kernel<<<dim3(SS / 64, BB * HH), 128, 0, stream>>>(q_bf, k_bf, v_bf, pad, at_bf);

  // 4) output projection (f32 out)
  gemm_bf16_kernel<false><<<gDD, 256, 0, stream>>>(at_bf, wo_bf, bo, proj_f, (int)MT, DD, DD);

  // 5) residual + LN1 -> h (f32 + bf16)
  add_layernorm_kernel<<<(unsigned)MT, 256, 0, stream>>>(hidden, proj_f, ln1w, ln1b, h_f, h_bf);

  // 6) FFN up (f32 out), GELU (bf16 out), FFN down (f32 out)
  const dim3 g4D(4 * DD / 128, (unsigned)(MT / 128));
  gemm_bf16_kernel<false><<<g4D, 256, 0, stream>>>(h_bf, w1_bf, b1, ff1_f, (int)MT, 4 * DD, DD);
  gelu_bf16_kernel<<<dim3((unsigned)((MT * 4 * DD + 1023) / 1024)), 256, 0, stream>>>(
      ff1_f, ff1_bf, MT * 4 * DD);
  gemm_bf16_kernel<false><<<gDD, 256, 0, stream>>>(ff1_bf, w2_bf, b2, ff2_f, (int)MT, DD, 4 * DD);

  // 7) residual + LN2 -> d_out (f32)
  add_layernorm_kernel<<<(unsigned)MT, 256, 0, stream>>>(h_f, ff2_f, ln2w, ln2b, outp, nullptr);
}
